// _GroundingDinoDecoder_72808285602310
// MI455X (gfx1250) — matrix operand order repaired and checked by reading, not compiled, not run on hardware
//
#include <hip/hip_runtime.h>

// ---------------------------------------------------------------------------
// GroundingDINO bi-directional cross-attention for MI455X (gfx1250, wave32)
// All GEMMs run on v_wmma_f32_16x16x32_bf16 matrix cores.
// ---------------------------------------------------------------------------

typedef __attribute__((ext_vector_type(16))) __bf16 v16bf;
typedef __attribute__((ext_vector_type(8)))  float  v8f;
typedef unsigned short u16;
typedef unsigned int   u32;

// problem constants
constexpr int NB   = 2;
constexpr int TGT  = 13600;
constexpr int SRC  = 256;
constexpr int VDIM = 256;
constexpr int TDIM = 256;
constexpr int EMB  = 1024;
constexpr int NH   = 4;
constexpr int HD   = 256;
constexpr int BH   = NB * NH;          // 8
constexpr float FP16MIN = -65504.0f;
constexpr float CLAMPV  = 50000.0f;
constexpr int TCH = 512;               // t-chunk for column softmax
constexpr int NCH = (TGT + TCH - 1) / TCH;   // 27

// gemm flags
#define GF_F32OUT 1
#define GF_TRANS  2
#define GF_REDMAX 4

__device__ __forceinline__ u16 f32_to_bf16(float f) {
  u32 x = __float_as_uint(f);
  u32 lsb = (x >> 16) & 1u;
  x += 0x7FFFu + lsb;                   // round-to-nearest-even
  return (u16)(x >> 16);
}
// order-preserving float->u32 key for deterministic atomic max
__device__ __forceinline__ u32 fkey(float f) {
  u32 b = __float_as_uint(f);
  return (b & 0x80000000u) ? ~b : (b | 0x80000000u);
}
__device__ __forceinline__ float funkey(u32 k) {
  u32 b = (k & 0x80000000u) ? (k & 0x7FFFFFFFu) : ~k;
  return __uint_as_float(b);
}
__device__ __forceinline__ float clampc(float v) {
  return fminf(fmaxf(v, -CLAMPV), CLAMPV);
}

union Frag { uint4 q[2]; v16bf v; };

// ---------------------------------------------------------------------------
// generic batched WMMA GEMM:  C[m,n] = scale * (sum_k A[m,k]*B[n,k] + bias[n])
// A: M x K bf16 row-major (lda).  B: N x K bf16 row-major (ldb) (i.e. B^T).
// batch z -> (zb = z/batchH, zh = z%batchH), per-operand 2-level strides.
// Block: 256 threads = 8 waves; each wave computes a 16(M) x 64(N) tile.
// ---------------------------------------------------------------------------
__global__ __launch_bounds__(256) void k_wmma_gemm(
    const u16* __restrict__ A, const u16* __restrict__ Bm,
    const float* __restrict__ bias, void* __restrict__ Cv,
    int M, int N, int K, int lda, int ldb, int ldc,
    long sA0, long sA1, long sB0, long sB1, long sC0, long sC1,
    int batchH, float scale, int flags, u32* gmaxk)
{
  const int z  = blockIdx.z;
  const int zb = z / batchH, zh = z % batchH;
  A  += (long)zb * sA0 + (long)zh * sA1;
  Bm += (long)zb * sB0 + (long)zh * sB1;
  const long coff = (long)zb * sC0 + (long)zh * sC1;

  const int wave = threadIdx.x >> 5;
  const int lane = threadIdx.x & 31;
  const int tileM = (blockIdx.x * 8 + wave) * 16;
  if (tileM >= M) return;                       // wave-uniform: EXEC stays full
  const int tileN = blockIdx.y * 64;
  const int l  = lane & 15;
  const int hi = lane >> 4;

  // ISA 16-bit A layout: lane<16 holds K = k0+[0..7]&[16..23]; lane>=16: +8
  const u16* arow  = A  + (long)(tileM + l) * lda + hi * 8;
  // ISA 16-bit B layout: lane<16 holds K = k0+[0..7]&[16..23]; lane>=16: +8
  const u16* brow0 = Bm + (long)(tileN + l) * ldb + hi * 8;
  const long bstep = (long)16 * ldb;

  v8f zero = {0.f,0.f,0.f,0.f,0.f,0.f,0.f,0.f};
  v8f acc[4] = {zero, zero, zero, zero};

  for (int k0 = 0; k0 < K; k0 += 32) {
    Frag fa;
    fa.q[0] = *(const uint4*)(arow + k0);
    fa.q[1] = *(const uint4*)(arow + k0 + 16);
#pragma unroll
    for (int j = 0; j < 4; ++j) {
      Frag fb;
      const u16* bp = brow0 + j * bstep + k0;
      fb.q[0] = *(const uint4*)(bp);
      fb.q[1] = *(const uint4*)(bp + 16);
      acc[j] = __builtin_amdgcn_wmma_f32_16x16x32_bf16(
          false, fa.v, false, fb.v, (short)0, acc[j], false, false);
    }
  }

  float wmax = -3.4e38f;
#pragma unroll
  for (int j = 0; j < 4; ++j) {
    const int n = tileN + j * 16 + l;
    const float bn = bias ? bias[n] : 0.f;
#pragma unroll
    for (int r = 0; r < 8; ++r) {
      const int m = tileM + r + hi * 8;       // ISA f32 C/D layout
      float v = (acc[j][r] + bn) * scale;
      if (flags & GF_REDMAX) wmax = fmaxf(wmax, v);
      if (flags & GF_F32OUT) {
        ((float*)Cv)[coff + (long)m * ldc + n] = v;
      } else if (flags & GF_TRANS) {
        ((u16*)Cv)[coff + (long)n * ldc + m] = f32_to_bf16(v);
      } else {
        ((u16*)Cv)[coff + (long)m * ldc + n] = f32_to_bf16(v);
      }
    }
  }
  if (flags & GF_REDMAX) {
#pragma unroll
    for (int o = 16; o; o >>= 1) wmax = fmaxf(wmax, __shfl_xor(wmax, o));
    if (lane == 0) atomicMax(gmaxk, fkey(wmax));
  }
}

// ---------------------------------------------------------------------------
// helpers
// ---------------------------------------------------------------------------
__global__ void k_cvt_bf16(const float* __restrict__ s, u16* __restrict__ d, int n) {
  int i = blockIdx.x * blockDim.x + threadIdx.x;
  if (i < n) d[i] = f32_to_bf16(s[i]);
}

__global__ void k_transpose_cvt(const float* __restrict__ W, u16* __restrict__ WT,
                                int K, int N) {          // W: KxN -> WT: NxK bf16
  __shared__ float tile[32][33];
  const int k0 = blockIdx.y * 32, n0 = blockIdx.x * 32;
  const int tx = threadIdx.x, ty = threadIdx.y;          // 32 x 8
  for (int i = ty; i < 32; i += 8) {
    int k = k0 + i, n = n0 + tx;
    tile[i][tx] = (k < K && n < N) ? W[(long)k * N + n] : 0.f;
  }
  __syncthreads();
  for (int i = ty; i < 32; i += 8) {
    int n = n0 + i, k = k0 + tx;
    if (n < N && k < K) WT[(long)n * K + k] = f32_to_bf16(tile[tx][i]);
  }
}

__global__ void k_init_gmax(u32* g) { *g = 0u; }

// vision softmax over SRC: one 256-thread block per (bh, t) row
__global__ void k_vision_softmax(const float* __restrict__ attn,
                                 const float* __restrict__ mask,
                                 const u32* __restrict__ gmaxk,
                                 u16* __restrict__ out) {
  const long row = blockIdx.x;                 // bh*TGT + t
  const int  s   = threadIdx.x;
  const int  bh  = (int)(row / TGT);
  const int  b   = bh / NH;
  const float gmax = funkey(*gmaxk);
  const float c = clampc(attn[row * SRC + s] - gmax);
  const float logit = c + mask[b * SRC + s] * FP16MIN;
  __shared__ float red[8];
  float m = logit;
#pragma unroll
  for (int o = 16; o; o >>= 1) m = fmaxf(m, __shfl_xor(m, o));
  if ((s & 31) == 0) red[s >> 5] = m;
  __syncthreads();
  float bm = red[0];
#pragma unroll
  for (int i = 1; i < 8; ++i) bm = fmaxf(bm, red[i]);
  const float e = __expf(logit - bm);
  float ss = e;
#pragma unroll
  for (int o = 16; o; o >>= 1) ss += __shfl_xor(ss, o);
  __syncthreads();
  if ((s & 31) == 0) red[s >> 5] = ss;
  __syncthreads();
  float tot = 0.f;
#pragma unroll
  for (int i = 0; i < 8; ++i) tot += red[i];
  out[row * SRC + s] = f32_to_bf16(e / tot);
}

// text softmax over TGT (columns of attn), deterministic chunked reduction
__global__ void k_col_partial_max(const float* __restrict__ attn,
                                  const u32* __restrict__ gmaxk,
                                  float* __restrict__ pmax) {
  const int bh = blockIdx.x, ch = blockIdx.y, s = threadIdx.x;
  const float gmax = funkey(*gmaxk);
  const float* A = attn + (long)bh * TGT * SRC;
  const int t0 = ch * TCH, t1 = min(t0 + TCH, TGT);
  float m = -3.4e38f;
  for (int t = t0; t < t1; ++t) m = fmaxf(m, clampc(A[(long)t * SRC + s] - gmax));
  pmax[((long)bh * NCH + ch) * SRC + s] = m;
}
__global__ void k_col_reduce(const float* __restrict__ part, float* __restrict__ out,
                             int mode) {                  // 0=max, 1=sum
  const int bh = blockIdx.x, s = threadIdx.x;
  float acc = mode ? 0.f : -3.4e38f;
  for (int c = 0; c < NCH; ++c) {
    float v = part[((long)bh * NCH + c) * SRC + s];
    acc = mode ? (acc + v) : fmaxf(acc, v);
  }
  out[bh * SRC + s] = acc;
}
__global__ void k_col_partial_sum(const float* __restrict__ attn,
                                  const u32* __restrict__ gmaxk,
                                  const float* __restrict__ colmax,
                                  float* __restrict__ psum) {
  const int bh = blockIdx.x, ch = blockIdx.y, s = threadIdx.x;
  const float gmax = funkey(*gmaxk);
  const float cm = colmax[bh * SRC + s];
  const float* A = attn + (long)bh * TGT * SRC;
  const int t0 = ch * TCH, t1 = min(t0 + TCH, TGT);
  float ss = 0.f;
  for (int t = t0; t < t1; ++t)
    ss += __expf(clampc(clampc(A[(long)t * SRC + s] - gmax) - cm));
  psum[((long)bh * NCH + ch) * SRC + s] = ss;
}
__global__ void k_text_weights(const float* __restrict__ attn,
                               const u32* __restrict__ gmaxk,
                               const float* __restrict__ colmax,
                               const float* __restrict__ colsum,
                               u16* __restrict__ out) {  // (bh, SRC, TGT) bf16
  const int bh = blockIdx.x, ch = blockIdx.y, s = threadIdx.x;
  const float gmax = funkey(*gmaxk);
  const float cm  = colmax[bh * SRC + s];
  const float inv = 1.f / colsum[bh * SRC + s];
  const float* A = attn + (long)bh * TGT * SRC;
  u16* O = out + ((long)bh * SRC + s) * TGT;
  const int t0 = ch * TCH, t1 = min(t0 + TCH, TGT);
  for (int t = t0; t < t1; ++t) {
    float e = __expf(clampc(clampc(A[(long)t * SRC + s] - gmax) - cm));
    O[t] = f32_to_bf16(e * inv);
  }
}

// ---------------------------------------------------------------------------
static inline void gemm(hipStream_t st, const u16* A, const u16* B, const float* bias,
                        void* C, int M, int N, int K, int lda, int ldb, int ldc,
                        long sA0, long sA1, long sB0, long sB1, long sC0, long sC1,
                        int nb0, int batchH, float scale, int flags, u32* gmaxk) {
  dim3 g((unsigned)((M + 127) / 128), (unsigned)(N / 64), (unsigned)(nb0 * batchH));
  k_wmma_gemm<<<g, dim3(256), 0, st>>>(A, B, bias, C, M, N, K, lda, ldb, ldc,
                                       sA0, sA1, sB0, sB1, sC0, sC1,
                                       batchH, scale, flags, gmaxk);
}

extern "C" void kernel_launch(void* const* d_in, const int* in_sizes, int n_in,
                              void* d_out, int out_size, void* d_ws, size_t ws_size,
                              hipStream_t stream) {
  (void)in_sizes; (void)n_in; (void)out_size; (void)ws_size;
  const float* vis  = (const float*)d_in[0];
  const float* txt  = (const float*)d_in[1];
  const float* mask = (const float*)d_in[2];
  const float* Wvq = (const float*)d_in[3],  *bvq = (const float*)d_in[4];
  const float* Wtk = (const float*)d_in[5],  *btk = (const float*)d_in[6];
  const float* Wvv = (const float*)d_in[7],  *bvv = (const float*)d_in[8];
  const float* Wtv = (const float*)d_in[9],  *btv = (const float*)d_in[10];
  const float* Wvo = (const float*)d_in[11], *bvo = (const float*)d_in[12];
  const float* Wto = (const float*)d_in[13], *bto = (const float*)d_in[14];

  // ---- workspace carve-out (~298 MB total) ----
  unsigned char* w = (unsigned char*)d_ws;
  size_t off = 0;
  auto alloc = [&](size_t bytes) -> void* {
    off = (off + 255) & ~(size_t)255;
    void* p = w + off; off += bytes; return p;
  };
  u16* visB  = (u16*)alloc((size_t)NB * TGT * VDIM * 2);
  u16* txtB  = (u16*)alloc((size_t)NB * SRC * TDIM * 2);
  u16* WqT   = (u16*)alloc((size_t)EMB * VDIM * 2);
  u16* WkT   = (u16*)alloc((size_t)EMB * TDIM * 2);
  u16* WvvT  = (u16*)alloc((size_t)EMB * VDIM * 2);
  u16* WtvT  = (u16*)alloc((size_t)EMB * TDIM * 2);
  u16* WoVT  = (u16*)alloc((size_t)VDIM * EMB * 2);
  u16* WoTT  = (u16*)alloc((size_t)TDIM * EMB * 2);
  u16* qB    = (u16*)alloc((size_t)NB * TGT * EMB * 2);     // later: vision_attn
  u16* kB    = (u16*)alloc((size_t)NB * SRC * EMB * 2);
  u16* tvT   = (u16*)alloc((size_t)NB * EMB * SRC * 2);     // (bh, HD, SRC)
  u16* vvT   = (u16*)alloc((size_t)NB * EMB * TGT * 2);     // (bh, HD, TGT)
  float* attn = (float*)alloc((size_t)BH * TGT * SRC * 4);  // later: concat bf16
  u16* textAttn = (u16*)alloc((size_t)BH * SRC * TGT * 2);
  u16* tconcat  = (u16*)alloc((size_t)NB * SRC * EMB * 2);
  float* pmax   = (float*)alloc((size_t)BH * NCH * SRC * 4);
  float* psum   = (float*)alloc((size_t)BH * NCH * SRC * 4);
  float* colmax = (float*)alloc((size_t)BH * SRC * 4);
  float* colsum = (float*)alloc((size_t)BH * SRC * 4);
  u32* gmaxk    = (u32*)alloc(4);
  u16* visAttn  = qB;                       // reuse q after attn GEMM
  u16* concat   = (u16*)attn;               // reuse attn after softmaxes

  // ---- stage 0: conversions / weight transposes ----
  {
    int n = NB * TGT * VDIM;
    k_cvt_bf16<<<(n + 255) / 256, 256, 0, stream>>>(vis, visB, n);
    n = NB * SRC * TDIM;
    k_cvt_bf16<<<(n + 255) / 256, 256, 0, stream>>>(txt, txtB, n);
    dim3 blk(32, 8);
    k_transpose_cvt<<<dim3(EMB / 32, VDIM / 32), blk, 0, stream>>>(Wvq, WqT,  VDIM, EMB);
    k_transpose_cvt<<<dim3(EMB / 32, TDIM / 32), blk, 0, stream>>>(Wtk, WkT,  TDIM, EMB);
    k_transpose_cvt<<<dim3(EMB / 32, VDIM / 32), blk, 0, stream>>>(Wvv, WvvT, VDIM, EMB);
    k_transpose_cvt<<<dim3(EMB / 32, TDIM / 32), blk, 0, stream>>>(Wtv, WtvT, TDIM, EMB);
    k_transpose_cvt<<<dim3(VDIM / 32, EMB / 32), blk, 0, stream>>>(Wvo, WoVT, EMB, VDIM);
    k_transpose_cvt<<<dim3(TDIM / 32, EMB / 32), blk, 0, stream>>>(Wto, WoTT, EMB, TDIM);
    k_init_gmax<<<1, 1, 0, stream>>>(gmaxk);
  }

  const float scaleQ = 0.0625f;            // HD^-0.5 = 1/16

  // ---- stage 1: projections (WMMA) ----
  // q = (vis @ Wvq + bvq) * scale  -> (B,TGT,EMB) bf16
  gemm(stream, visB, WqT, bvq, qB, NB * TGT, EMB, VDIM, VDIM, VDIM, EMB,
       0, 0, 0, 0, 0, 0, 1, 1, scaleQ, 0, nullptr);
  // k = txt @ Wtk + btk -> (B,SRC,EMB) bf16
  gemm(stream, txtB, WkT, btk, kB, NB * SRC, EMB, TDIM, TDIM, TDIM, EMB,
       0, 0, 0, 0, 0, 0, 1, 1, 1.f, 0, nullptr);
  // tv^T: per-b GEMM, transposed store -> (bh, HD, SRC)
  gemm(stream, txtB, WtvT, btv, tvT, SRC, EMB, TDIM, TDIM, TDIM, SRC,
       (long)SRC * TDIM, 0, 0, 0, (long)EMB * SRC, 0, NB, 1, 1.f, GF_TRANS, nullptr);
  // vv^T: per-b GEMM, transposed store -> (bh, HD, TGT)
  gemm(stream, visB, WvvT, bvv, vvT, TGT, EMB, VDIM, VDIM, VDIM, TGT,
       (long)TGT * VDIM, 0, 0, 0, (long)EMB * TGT, 0, NB, 1, 1.f, GF_TRANS, nullptr);

  // ---- stage 2: attn = q . k^T, fused global max (WMMA, f32 out) ----
  gemm(stream, qB, kB, nullptr, attn, TGT, SRC, HD, EMB, EMB, SRC,
       (long)TGT * EMB, HD, (long)SRC * EMB, HD,
       (long)NH * TGT * SRC, (long)TGT * SRC,
       NB, NH, 1.f, GF_F32OUT | GF_REDMAX, gmaxk);

  // ---- stage 3: softmaxes ----
  k_vision_softmax<<<BH * TGT, SRC, 0, stream>>>(attn, mask, gmaxk, visAttn);
  k_col_partial_max<<<dim3(BH, NCH), SRC, 0, stream>>>(attn, gmaxk, pmax);
  k_col_reduce<<<BH, SRC, 0, stream>>>(pmax, colmax, 0);
  k_col_partial_sum<<<dim3(BH, NCH), SRC, 0, stream>>>(attn, gmaxk, colmax, psum);
  k_col_reduce<<<BH, SRC, 0, stream>>>(psum, colsum, 1);
  k_text_weights<<<dim3(BH, NCH), SRC, 0, stream>>>(attn, gmaxk, colmax, colsum, textAttn);

  // ---- stage 4: attention-weighted values (WMMA) ----
  // concat = vision_attn @ tv   -> (B,TGT,EMB) bf16  (reuses attn buffer)
  gemm(stream, visAttn, tvT, nullptr, concat, TGT, HD, SRC, SRC, SRC, EMB,
       (long)NH * TGT * SRC, (long)TGT * SRC, (long)NH * HD * SRC, (long)HD * SRC,
       (long)TGT * EMB, HD, NB, NH, 1.f, 0, nullptr);
  // tconcat = text_attn @ vv    -> (B,SRC,EMB) bf16
  gemm(stream, textAttn, vvT, nullptr, tconcat, SRC, HD, TGT, TGT, TGT, EMB,
       (long)NH * SRC * TGT, (long)SRC * TGT, (long)NH * HD * TGT, (long)HD * TGT,
       (long)SRC * EMB, HD, NB, NH, 1.f, 0, nullptr);

  // ---- stage 5: output projections (WMMA, f32 stores into d_out) ----
  float* visOut = (float*)d_out;
  float* txtOut = visOut + (size_t)NB * TGT * VDIM;
  gemm(stream, concat, WoVT, bvo, visOut, NB * TGT, VDIM, EMB, EMB, EMB, VDIM,
       0, 0, 0, 0, 0, 0, 1, 1, 1.f, GF_F32OUT, nullptr);
  gemm(stream, tconcat, WoTT, bto, txtOut, NB * SRC, TDIM, EMB, EMB, EMB, TDIM,
       0, 0, 0, 0, 0, 0, 1, 1, 1.f, GF_F32OUT, nullptr);
}
